// DAHead_69707319214291
// MI455X (gfx1250) — matrix operand order repaired and checked by reading, not compiled, not run on hardware
//
#include <hip/hip_runtime.h>
#include <hip/hip_bf16.h>

// ---------------------------------------------------------------------------
// DANet dual-attention head for MI455X (gfx1250), wave32 + WMMA bf16.
// B=4, C=512, Ci=256, N=4096.  All heavy math runs through
// v_wmma_f32_16x16x32_bf16 (f32 accumulation); softmax in f32.
// ---------------------------------------------------------------------------

typedef __bf16 bf16_t;
typedef __attribute__((ext_vector_type(16))) __bf16 v16bf;
typedef __attribute__((ext_vector_type(8)))  float  v8f;
typedef __attribute__((ext_vector_type(4)))  float  v4f;

#define BQ 4
#define CC 512
#define CI 256
#define NN 4096

// ---- WMMA fragment loaders (CDNA5 ISA §7.12.2 layouts, wave32) -------------
// A: 16x32 bf16 tile from row-major [M,K]; lane = M row, lane-half selects
// K groups {0-7,16-23} vs {8-15,24-31}.
static __device__ __forceinline__ v16bf frag_a(const bf16_t* __restrict__ A,
                                               int lda, int m0, int k0, int lane) {
  const bf16_t* p = A + (size_t)(m0 + (lane & 15)) * lda + k0 + (lane >> 4) * 8;
  union { v16bf v; struct { v4f lo, hi; } q; } u;
  u.q.lo = *(const v4f*)(p);
  u.q.hi = *(const v4f*)(p + 16);
  return u.v;
}
// B: 32x16 bf16 tile loaded from row-major B^T [N,K]; lane = N column,
// lane-half selects K groups {0-7,16-23} vs {8-15,24-31} (matches frag_a).
static __device__ __forceinline__ v16bf frag_b(const bf16_t* __restrict__ BT,
                                               int ldb, int n0, int k0, int lane) {
  const bf16_t* p = BT + (size_t)(n0 + (lane & 15)) * ldb + k0 + (lane >> 4) * 8;
  union { v16bf v; struct { v4f lo, hi; } q; } u;
  u.q.lo = *(const v4f*)(p);
  u.q.hi = *(const v4f*)(p + 16);
  return u.v;
}

// ---- Generic bf16 GEMM: out[M,Nn] = A[M,K] * BT[Nn,K]^T, f32 accum ---------
// One wave per 16x64 strip (4 C fragments, A fragment reused 4x).
// Epilogues:
//   0: store f32                       (energy / cam-energy)
//   1: +bias[col]  -> bf16             (q/k projections, output transposed)
//   2: +bias[row]  -> bf16             (v projection)
//   3: sa = tanh(g*acc + x) -> f32 + bf16 [C,N] + bf16 [N,C]   (PAM fuse)
//   4: out = g*acc + resid  -> f32     (CAM output, straight to d_out)
struct GemmArgs {
  const bf16_t* A;  const bf16_t* BT;
  int lda, ldb;
  int M, Nn, K;
  int epi;
  float*  outf;  int ldo;
  bf16_t* outb;  int ldob;
  bf16_t* outbT; int ldobT;
  float*  outf2;
  const float* bias;
  const float* resid; int ldr;
  const float* gptr;
};

__global__ __launch_bounds__(128) void gemm_bf16_wmma(GemmArgs g) {
  const int lane = threadIdx.x & 31;
  const int wave = threadIdx.x >> 5;
  const int m0 = blockIdx.y * 16;
  const int n0 = (blockIdx.x * 4 + wave) * 64;
  if (m0 >= g.M || n0 >= g.Nn) return;   // wave-uniform; EXEC stays all-1s

  v8f acc0 = {0,0,0,0,0,0,0,0};
  v8f acc1 = {0,0,0,0,0,0,0,0};
  v8f acc2 = {0,0,0,0,0,0,0,0};
  v8f acc3 = {0,0,0,0,0,0,0,0};

  for (int k = 0; k < g.K; k += 32) {
    v16bf a  = frag_a(g.A,  g.lda, m0, k, lane);
    v16bf b0 = frag_b(g.BT, g.ldb, n0,      k, lane);
    v16bf b1 = frag_b(g.BT, g.ldb, n0 + 16, k, lane);
    v16bf b2 = frag_b(g.BT, g.ldb, n0 + 32, k, lane);
    v16bf b3 = frag_b(g.BT, g.ldb, n0 + 48, k, lane);
    acc0 = __builtin_amdgcn_wmma_f32_16x16x32_bf16(false, a, false, b0, (short)0, acc0, false, false);
    acc1 = __builtin_amdgcn_wmma_f32_16x16x32_bf16(false, a, false, b1, (short)0, acc1, false, false);
    acc2 = __builtin_amdgcn_wmma_f32_16x16x32_bf16(false, a, false, b2, (short)0, acc2, false, false);
    acc3 = __builtin_amdgcn_wmma_f32_16x16x32_bf16(false, a, false, b3, (short)0, acc3, false, false);
  }

  const float gval = g.gptr ? g.gptr[0] : 0.f;
  const int half = lane >> 4;
  const int nc   = lane & 15;
  v8f accs[4] = {acc0, acc1, acc2, acc3};

#pragma unroll
  for (int t = 0; t < 4; ++t) {
    const int n = n0 + t * 16 + nc;
    v8f acc = accs[t];
#pragma unroll
    for (int r = 0; r < 8; ++r) {
      const int m = m0 + r + half * 8;     // C/D layout: VGPR r -> row r (+8)
      const float vv = acc[r];
      if (g.epi == 0) {
        g.outf[(size_t)m * g.ldo + n] = vv;
      } else if (g.epi == 1) {
        g.outb[(size_t)m * g.ldob + n] = (bf16_t)(vv + g.bias[n]);
      } else if (g.epi == 2) {
        g.outb[(size_t)m * g.ldob + n] = (bf16_t)(vv + g.bias[m]);
      } else if (g.epi == 3) {
        const float s = tanhf(gval * vv + g.resid[(size_t)m * g.ldr + n]);
        g.outf2[(size_t)m * g.ldr + n]   = s;
        g.outb [(size_t)m * g.ldob + n]  = (bf16_t)s;
        g.outbT[(size_t)n * g.ldobT + m] = (bf16_t)s;
      } else {
        g.outf[(size_t)m * g.ldo + n] =
            gval * vv + g.resid[(size_t)m * g.ldr + n];
      }
    }
  }
}

// ---- f32 -> bf16 convert ---------------------------------------------------
__global__ void cvt_f32_to_bf16(const float* __restrict__ in,
                                bf16_t* __restrict__ out, int n) {
  int i = blockIdx.x * blockDim.x + threadIdx.x;
  if (i < n) out[i] = (bf16_t)in[i];
}

// ---- x [B,C,N] f32 -> xT [B,N,C] bf16 --------------------------------------
__global__ void xpose_bf16(const float* __restrict__ x, bf16_t* __restrict__ xT) {
  const int b   = blockIdx.z;
  const int idx = blockIdx.x * blockDim.x + threadIdx.x;   // < C*N (exact grid)
  const int c = idx / NN, n = idx % NN;
  xT[(size_t)b * NN * CC + (size_t)n * CC + c] =
      (bf16_t)x[(size_t)b * CC * NN + idx];
}

// ---- PAM row softmax: f32 energy row -> bf16 attn row, in place (aliased) --
// attn row i lives at the same byte offset as energy row i (bf16 ld = 2*NN).
__global__ __launch_bounds__(256) void softmax_pam_kernel(float* __restrict__ energy,
                                                          float scale) {
  const int row = blockIdx.x;
  float*  e = energy + (size_t)row * NN;
  bf16_t* a = reinterpret_cast<bf16_t*>(e);
  const int t = threadIdx.x;
  float v[16];
  float mx = -3.402823466e38f;
#pragma unroll
  for (int i = 0; i < 16; ++i) { v[i] = e[t + (i << 8)] * scale; mx = fmaxf(mx, v[i]); }
  __shared__ float red[256];
  red[t] = mx; __syncthreads();
  for (int s = 128; s > 0; s >>= 1) { if (t < s) red[t] = fmaxf(red[t], red[t + s]); __syncthreads(); }
  mx = red[0]; __syncthreads();
  float sm = 0.f;
#pragma unroll
  for (int i = 0; i < 16; ++i) { v[i] = __expf(v[i] - mx); sm += v[i]; }
  red[t] = sm; __syncthreads();
  for (int s = 128; s > 0; s >>= 1) { if (t < s) red[t] += red[t + s]; __syncthreads(); }
  const float inv = 1.f / (red[0] * (float)NN);
#pragma unroll
  for (int i = 0; i < 16; ++i) a[t + (i << 8)] = (bf16_t)(v[i] * inv);
}

// ---- CAM softmax: softmax(max - e) == exp(min - e)/sum ---------------------
__global__ __launch_bounds__(128) void softmax_cam_kernel(const float* __restrict__ ec,
                                                          bf16_t* __restrict__ ac,
                                                          float scale) {
  const int row = blockIdx.x;                 // b*C + c
  const float* e = ec + (size_t)row * CC;
  bf16_t*      a = ac + (size_t)row * CC;
  const int t = threadIdx.x;
  float v[4];
  float mn = 3.402823466e38f;
#pragma unroll
  for (int i = 0; i < 4; ++i) { v[i] = e[t + (i << 7)] * scale; mn = fminf(mn, v[i]); }
  __shared__ float red[128];
  red[t] = mn; __syncthreads();
  for (int s = 64; s > 0; s >>= 1) { if (t < s) red[t] = fminf(red[t], red[t + s]); __syncthreads(); }
  mn = red[0]; __syncthreads();
  float sm = 0.f;
#pragma unroll
  for (int i = 0; i < 4; ++i) { v[i] = __expf(mn - v[i]); sm += v[i]; }
  red[t] = sm; __syncthreads();
  for (int s = 64; s > 0; s >>= 1) { if (t < s) red[t] += red[t + s]; __syncthreads(); }
  const float inv = 1.f / (red[0] * (float)CC);
#pragma unroll
  for (int i = 0; i < 4; ++i) a[t + (i << 7)] = (bf16_t)(v[i] * inv);
}

// ---------------------------------------------------------------------------
extern "C" void kernel_launch(void* const* d_in, const int* in_sizes, int n_in,
                              void* d_out, int out_size, void* d_ws, size_t ws_size,
                              hipStream_t stream) {
  const float* x    = (const float*)d_in[0];
  const float* wq   = (const float*)d_in[1];
  const float* bq   = (const float*)d_in[2];
  const float* wk   = (const float*)d_in[3];
  const float* bk   = (const float*)d_in[4];
  const float* wv   = (const float*)d_in[5];
  const float* bv   = (const float*)d_in[6];
  const float* gpam = (const float*)d_in[7];
  const float* gcam = (const float*)d_in[8];
  float* out = (float*)d_out;
  (void)in_sizes; (void)n_in; (void)out_size; (void)ws_size;

  // ---- workspace carve-up (all offsets 256B aligned) ----
  char* ws = (char*)d_ws;
  size_t off = 0;
  auto carve = [&](size_t bytes) -> char* {
    char* p = ws + off;
    off = (off + bytes + 255) & ~(size_t)255;
    return p;
  };
  bf16_t* xT   = (bf16_t*)carve((size_t)BQ * NN * CC * 2);   // [B,N,C]
  bf16_t* wqb  = (bf16_t*)carve((size_t)CI * CC * 2);
  bf16_t* wkb  = (bf16_t*)carve((size_t)CI * CC * 2);
  bf16_t* wvb  = (bf16_t*)carve((size_t)CC * CC * 2);
  bf16_t* qT   = (bf16_t*)carve((size_t)BQ * NN * CI * 2);   // [B,N,Ci]
  bf16_t* kT   = (bf16_t*)carve((size_t)BQ * NN * CI * 2);   // [B,N,Ci]
  bf16_t* vB   = (bf16_t*)carve((size_t)BQ * CC * NN * 2);   // [B,C,N]
  float*  saf  = (float*) carve((size_t)BQ * CC * NN * 4);   // sa f32 [B,C,N]
  bf16_t* sab  = (bf16_t*)carve((size_t)BQ * CC * NN * 2);   // sa bf16 [B,C,N]
  bf16_t* sabT = (bf16_t*)carve((size_t)BQ * NN * CC * 2);   // sa bf16 [B,N,C]
  float*  ec   = (float*) carve((size_t)BQ * CC * CC * 4);   // cam energy
  bf16_t* ac   = (bf16_t*)carve((size_t)BQ * CC * CC * 2);   // cam attn
  float*  energy = (float*)carve((size_t)NN * NN * 4);       // per-batch, reused

  const float scale_pam = 0.04419417382f;  // 1/sqrt(C=512)
  const float scale_cam = 0.015625f;       // 1/sqrt(N=4096)

  auto launch_gemm = [&](GemmArgs g) {
    dim3 grid((g.Nn + 255) / 256, (g.M + 15) / 16, 1);
    gemm_bf16_wmma<<<grid, 128, 0, stream>>>(g);
  };
  auto zargs = []() { GemmArgs g; __builtin_memset(&g, 0, sizeof(g)); return g; };

  // 1) weights -> bf16
  cvt_f32_to_bf16<<<(CI * CC + 255) / 256, 256, 0, stream>>>(wq, wqb, CI * CC);
  cvt_f32_to_bf16<<<(CI * CC + 255) / 256, 256, 0, stream>>>(wk, wkb, CI * CC);
  cvt_f32_to_bf16<<<(CC * CC + 255) / 256, 256, 0, stream>>>(wv, wvb, CC * CC);

  // 2) x -> xT bf16 [B,N,C]
  xpose_bf16<<<dim3(CC * NN / 256, 1, BQ), 256, 0, stream>>>(x, xT);

  // 3) projections (per batch)
  for (int b = 0; b < BQ; ++b) {
    const bf16_t* xT_b = xT + (size_t)b * NN * CC;
    // qT[n,o] = sum_c xT[n,c] wq[o,c] + bq[o]
    GemmArgs gq = zargs();
    gq.A = xT_b; gq.lda = CC; gq.BT = wqb; gq.ldb = CC;
    gq.M = NN; gq.Nn = CI; gq.K = CC;
    gq.epi = 1; gq.outb = qT + (size_t)b * NN * CI; gq.ldob = CI; gq.bias = bq;
    launch_gemm(gq);
    GemmArgs gk = gq;
    gk.BT = wkb; gk.outb = kT + (size_t)b * NN * CI; gk.bias = bk;
    launch_gemm(gk);
    // v[o,n] = sum_c wv[o,c] xT[n,c] + bv[o]
    GemmArgs gv = zargs();
    gv.A = wvb; gv.lda = CC; gv.BT = xT_b; gv.ldb = CC;
    gv.M = CC; gv.Nn = NN; gv.K = CC;
    gv.epi = 2; gv.outb = vB + (size_t)b * CC * NN; gv.ldob = NN; gv.bias = bv;
    launch_gemm(gv);
  }

  // 4) PAM per batch (energy buffer reused; attn bf16 aliases energy rows)
  for (int b = 0; b < BQ; ++b) {
    const bf16_t* qT_b = qT + (size_t)b * NN * CI;
    const bf16_t* kT_b = kT + (size_t)b * NN * CI;
    GemmArgs ge = zargs();
    ge.A = qT_b; ge.lda = CI; ge.BT = kT_b; ge.ldb = CI;
    ge.M = NN; ge.Nn = NN; ge.K = CI;
    ge.epi = 0; ge.outf = energy; ge.ldo = NN;
    launch_gemm(ge);

    softmax_pam_kernel<<<NN, 256, 0, stream>>>(energy, scale_pam);

    // pam_out[c,i] = sum_j v[c,j] attn[i,j]; fused sa = tanh(g*pam + x)
    GemmArgs gp = zargs();
    gp.A = vB + (size_t)b * CC * NN; gp.lda = NN;
    gp.BT = (const bf16_t*)energy;  gp.ldb = 2 * NN;   // aliased bf16 attn
    gp.M = CC; gp.Nn = NN; gp.K = NN;
    gp.epi = 3;
    gp.outf2 = saf + (size_t)b * CC * NN;
    gp.outb  = sab + (size_t)b * CC * NN;  gp.ldob  = NN;
    gp.outbT = sabT + (size_t)b * NN * CC; gp.ldobT = CC;
    gp.resid = x + (size_t)b * CC * NN;    gp.ldr   = NN;
    gp.gptr  = gpam;
    launch_gemm(gp);
  }

  // 5) CAM: Gram matrix, softmax, output
  for (int b = 0; b < BQ; ++b) {
    const bf16_t* sab_b = sab + (size_t)b * CC * NN;
    GemmArgs gc = zargs();
    gc.A = sab_b; gc.lda = NN; gc.BT = sab_b; gc.ldb = NN;
    gc.M = CC; gc.Nn = CC; gc.K = NN;
    gc.epi = 0; gc.outf = ec + (size_t)b * CC * CC; gc.ldo = CC;
    launch_gemm(gc);
  }
  softmax_cam_kernel<<<BQ * CC, 128, 0, stream>>>(ec, ac, scale_cam);
  for (int b = 0; b < BQ; ++b) {
    // out[c,n] = gcam * sum_d ac[c,d] sa[d,n] + sa[c,n]
    GemmArgs go = zargs();
    go.A = ac + (size_t)b * CC * CC; go.lda = CC;
    go.BT = sabT + (size_t)b * NN * CC; go.ldb = CC;
    go.M = CC; go.Nn = NN; go.K = CC;
    go.epi = 4;
    go.outf = out + (size_t)b * CC * NN; go.ldo = NN;
    go.resid = saf + (size_t)b * CC * NN; go.ldr = NN;
    go.gptr = gcam;
    launch_gemm(go);
  }
}
